// EdgeConvBlock_36240934043761
// MI455X (gfx1250) — matrix operand order repaired and checked by reading, not compiled, not run on hardware
//
#include <hip/hip_runtime.h>

// ---------------------------------------------------------------------------
// EdgeConv block for MI455X (gfx1250, wave32, WMMA f16 16x16x32)
// ---------------------------------------------------------------------------

typedef __attribute__((ext_vector_type(16))) _Float16 v16h;
typedef __attribute__((ext_vector_type(8)))  _Float16 v8h;
typedef __attribute__((ext_vector_type(8)))  float    v8f;

#define NE      800000      // edges
#define NNODES  50000       // nodes
#define CH      64          // LAYER width
#define KIN     128         // 2*IN_SIZE (concat)
#define BN_EPS  1e-5f

// ---- workspace byte offsets -----------------------------------------------
#define WS_W0T   0u                       // 128*64 f16 = 16384 B
#define WS_W1T   16384u                   //  64*64 f16 =  8192 B
#define WS_W2T   24576u                   //  64*64 f16 =  8192 B
#define WS_STATS 32768u                   // 768 f32    =  3072 B
#define WS_CNT   36864u                   // NNODES f32 = 200000 B
#define WS_H0    (1u<<20)                 // NE*CH f16  = 102.4 MB
#define WS_H1    (WS_H0 + 104857600u)     // NE*CH f16  = 102.4 MB
// float offsets inside WS_STATS
#define ST_SUM0   0
#define ST_SQ0   64
#define ST_SUM1 128
#define ST_SQ1  192
#define ST_SUM2 256
#define ST_SQ2  320
#define ST_SC0  384
#define ST_SH0  448
#define ST_SC1  512
#define ST_SH1  576
#define ST_SC2  640
#define ST_SH2  704

// ---------------------------------------------------------------------------
// WMMA fragment loads from LDS (layouts per CDNA5 ISA 7.12.2, wave32)
// A 16x32 f16: lane L holds row M=L&15; halves [0..7]=K[kb..kb+7],
//              [8..15]=K[kb+16..kb+23], kb = (L>>4)*8  (+ k-step base)
// B 32x16 f16 (stored transposed [N][K] in LDS): lane L holds col N=L&15,
//              same K order as A: kb = (L>>4)*8 (+ k-step base)
// ---------------------------------------------------------------------------
__device__ inline v16h load_a_frag(const _Float16* p) {
  v8h lo = *(const v8h*)p;          // 16B aligned (row pad = 8 halves)
  v8h hi = *(const v8h*)(p + 16);
  v16h a;
#pragma unroll
  for (int i = 0; i < 8; ++i) { a[i] = lo[i]; a[i + 8] = hi[i]; }
  return a;
}
__device__ inline v16h load_b_frag(const _Float16* p) {
  v8h lo = *(const v8h*)p;
  v8h hi = *(const v8h*)(p + 16);
  v16h b;
#pragma unroll
  for (int i = 0; i < 8; ++i) { b[i] = lo[i]; b[i + 8] = hi[i]; }
  return b;
}

// ---------------------------------------------------------------------------
// 1) init: zero accumulators, convert weights to f16 transposed [N][K]
// ---------------------------------------------------------------------------
__global__ void k_init(const float* __restrict__ W0, const float* __restrict__ W1,
                       const float* __restrict__ W2, float* __restrict__ out_sum,
                       unsigned char* __restrict__ ws) {
  _Float16* w0t = (_Float16*)(ws + WS_W0T);
  _Float16* w1t = (_Float16*)(ws + WS_W1T);
  _Float16* w2t = (_Float16*)(ws + WS_W2T);
  float* stats  = (float*)(ws + WS_STATS);
  float* counts = (float*)(ws + WS_CNT);
  const long long stride = (long long)gridDim.x * blockDim.x;
  const long long t0 = (long long)blockIdx.x * blockDim.x + threadIdx.x;
  for (long long i = t0; i < (long long)NNODES * CH; i += stride) out_sum[i] = 0.0f;
  for (long long i = t0; i < NNODES; i += stride) counts[i] = 0.0f;
  for (long long i = t0; i < 768; i += stride) stats[i] = 0.0f;
  for (long long i = t0; i < (long long)KIN * CH; i += stride) {
    const int k = (int)(i >> 6), n = (int)(i & 63);      // W0 row-major [K][N]
    w0t[n * KIN + k] = (_Float16)W0[i];
  }
  for (long long i = t0; i < (long long)CH * CH; i += stride) {
    const int k = (int)(i >> 6), n = (int)(i & 63);
    w1t[n * CH + k] = (_Float16)W1[i];
    w2t[n * CH + k] = (_Float16)W2[i];
  }
}

// ---------------------------------------------------------------------------
// 2) layer0: gather concat(x[dst],x[src]) -> f16 GEMM (K=128) + bias,
//    store h0 (f16), accumulate BN sum/sumsq (LDS reduce -> global atomics)
//    64 edges / block, 4 waves, each wave: 16 edges x 64 outputs
// ---------------------------------------------------------------------------
__global__ __launch_bounds__(128)
void k_edge_gemm0(const float* __restrict__ x, const long long* __restrict__ ei,
                  const float* __restrict__ b0, unsigned char* __restrict__ ws) {
  __shared__ _Float16 lA[64][KIN + 8];
  __shared__ _Float16 lW[CH][KIN + 8];
  __shared__ float lsum[CH], lsq[CH];
  const _Float16* w0t = (const _Float16*)(ws + WS_W0T);
  _Float16* h0 = (_Float16*)(ws + WS_H0);
  float* gsum = (float*)(ws + WS_STATS) + ST_SUM0;
  float* gsq  = (float*)(ws + WS_STATS) + ST_SQ0;

  const int tid = threadIdx.x;
  if (tid < CH) { lsum[tid] = 0.0f; lsq[tid] = 0.0f; }
  for (int i = tid; i < CH * KIN; i += 128) lW[i >> 7][i & 127] = w0t[i];

  const long long e0 = (long long)blockIdx.x * 64;
  {                                   // gather: 128 tasks = (edge, half)
    const int e_loc = tid >> 1, half = tid & 1;
    const long long eg = e0 + e_loc;
    // concat order: x_i = x[dst] first (rows 0..63 of W0), x_j = x[src]
    const long long node = half ? ei[eg] : ei[(long long)NE + eg];
    const float* xr = x + node * 64;
#pragma unroll
    for (int c = 0; c < 64; c += 4) {
      const float4 v = *(const float4*)(xr + c);
      lA[e_loc][half * 64 + c + 0] = (_Float16)v.x;
      lA[e_loc][half * 64 + c + 1] = (_Float16)v.y;
      lA[e_loc][half * 64 + c + 2] = (_Float16)v.z;
      lA[e_loc][half * 64 + c + 3] = (_Float16)v.w;
    }
  }
  __syncthreads();

  const int lane  = tid & 31, wave = tid >> 5;
  const int m0    = wave * 16;
  const int arow  = m0 + (lane & 15);
  const int akoff = (lane >> 4) << 3;   // 0 / 8
  const int bkoff = (lane >> 4) << 3;   // 0 / 8
  const int ncol  = lane & 15;
  const int mr0   = (lane >> 4) << 3;   // D rows 0..7 / 8..15

  v16h afr[4];
#pragma unroll
  for (int kk = 0; kk < 4; ++kk)
    afr[kk] = load_a_frag(&lA[arow][kk * 32 + akoff]);

#pragma unroll
  for (int n0 = 0; n0 < CH; n0 += 16) {
    v8f acc = {};
#pragma unroll
    for (int kk = 0; kk < 4; ++kk) {
      const v16h bf = load_b_frag(&lW[n0 + ncol][kk * 32 + bkoff]);
      acc = __builtin_amdgcn_wmma_f32_16x16x32_f16(false, afr[kk], false, bf,
                                                   (short)0, acc, false, false);
    }
    const int n = n0 + ncol;
    const float bias = b0[n];
    float s = 0.0f, q = 0.0f;
#pragma unroll
    for (int v = 0; v < 8; ++v) {
      const float val = acc[v] + bias;
      const long long eg = e0 + m0 + mr0 + v;
      h0[eg * CH + n] = (_Float16)val;
      s += val; q += val * val;
    }
    atomicAdd(&lsum[n], s);
    atomicAdd(&lsq[n], q);
  }
  __syncthreads();
  if (tid < CH) { atomicAdd(&gsum[tid], lsum[tid]); atomicAdd(&gsq[tid], lsq[tid]); }
}

// ---------------------------------------------------------------------------
// 3) fold BN stats into per-channel affine: scale = g*rsqrt(var+eps),
//    shift = beta - mu*scale
// ---------------------------------------------------------------------------
__global__ void k_bn_finalize(const float* __restrict__ gsum, const float* __restrict__ gsq,
                              const float* __restrict__ g, const float* __restrict__ beta,
                              float* __restrict__ scale, float* __restrict__ shift) {
  const int c = threadIdx.x;
  if (c < CH) {
    const float inv_e = 1.0f / (float)NE;
    const float mu  = gsum[c] * inv_e;
    const float var = gsq[c] * inv_e - mu * mu;
    const float sc  = g[c] * rsqrtf(var + BN_EPS);
    scale[c] = sc;
    shift[c] = beta[c] - mu * sc;
  }
}

// ---------------------------------------------------------------------------
// 4) mid layer: apply prev BN+ReLU while staging A, f16 GEMM (K=64) + bias,
//    store h_out (f16), accumulate next-layer BN stats
// ---------------------------------------------------------------------------
__global__ __launch_bounds__(128)
void k_edge_gemm_mid(const _Float16* __restrict__ h_in, _Float16* __restrict__ h_out,
                     const _Float16* __restrict__ wT, const float* __restrict__ bias,
                     const float* __restrict__ scale, const float* __restrict__ shift,
                     float* __restrict__ gsum, float* __restrict__ gsq) {
  __shared__ _Float16 lA[64][CH + 8];
  __shared__ _Float16 lW[CH][CH + 8];
  __shared__ float lsum[CH], lsq[CH], lscale[CH], lshift[CH];
  const int tid = threadIdx.x;
  if (tid < CH) {
    lsum[tid] = 0.0f; lsq[tid] = 0.0f;
    lscale[tid] = scale[tid]; lshift[tid] = shift[tid];
  }
  for (int i = tid; i < CH * CH; i += 128) lW[i >> 6][i & 63] = wT[i];
  __syncthreads();

  const long long e0 = (long long)blockIdx.x * 64;
  for (int i = tid; i < 64 * CH; i += 128) {         // coalesced f16 read
    const int c = i & 63;
    float v = (float)h_in[e0 * CH + i];
    v = fmaxf(v * lscale[c] + lshift[c], 0.0f);      // BN + ReLU on the fly
    lA[i >> 6][c] = (_Float16)v;
  }
  __syncthreads();

  const int lane  = tid & 31, wave = tid >> 5;
  const int m0    = wave * 16;
  const int arow  = m0 + (lane & 15);
  const int akoff = (lane >> 4) << 3;
  const int bkoff = (lane >> 4) << 3;
  const int ncol  = lane & 15;
  const int mr0   = (lane >> 4) << 3;

  v16h afr[2];
#pragma unroll
  for (int kk = 0; kk < 2; ++kk)
    afr[kk] = load_a_frag(&lA[arow][kk * 32 + akoff]);

#pragma unroll
  for (int n0 = 0; n0 < CH; n0 += 16) {
    v8f acc = {};
#pragma unroll
    for (int kk = 0; kk < 2; ++kk) {
      const v16h bf = load_b_frag(&lW[n0 + ncol][kk * 32 + bkoff]);
      acc = __builtin_amdgcn_wmma_f32_16x16x32_f16(false, afr[kk], false, bf,
                                                   (short)0, acc, false, false);
    }
    const int n = n0 + ncol;
    const float bb = bias[n];
    float s = 0.0f, q = 0.0f;
#pragma unroll
    for (int v = 0; v < 8; ++v) {
      const float val = acc[v] + bb;
      const long long eg = e0 + m0 + mr0 + v;
      h_out[eg * CH + n] = (_Float16)val;
      s += val; q += val * val;
    }
    atomicAdd(&lsum[n], s);
    atomicAdd(&lsq[n], q);
  }
  __syncthreads();
  if (tid < CH) { atomicAdd(&gsum[tid], lsum[tid]); atomicAdd(&gsq[tid], lsq[tid]); }
}

// ---------------------------------------------------------------------------
// 5) final BN+ReLU -> edge_activations (f32 output) + scatter-add to nodes
// ---------------------------------------------------------------------------
__global__ void k_edge_final(const _Float16* __restrict__ h2, const long long* __restrict__ ei,
                             const float* __restrict__ scale, const float* __restrict__ shift,
                             float* __restrict__ out_sum, float* __restrict__ edge_act,
                             float* __restrict__ counts) {
  const long long idx = (long long)blockIdx.x * blockDim.x + threadIdx.x;
  if (idx >= (long long)NE * CH) return;
  const int c = (int)(idx & 63);
  const long long e = idx >> 6;
  float v = (float)h2[idx];
  v = fmaxf(v * scale[c] + shift[c], 0.0f);
  edge_act[idx] = v;
  const long long dst = ei[(long long)NE + e];
  atomicAdd(&out_sum[dst * CH + c], v);
  if (c == 0) atomicAdd(&counts[dst], 1.0f);
}

// ---------------------------------------------------------------------------
// 6) mean: out = sum / max(count, 1)
// ---------------------------------------------------------------------------
__global__ void k_node_div(float* __restrict__ out_sum, const float* __restrict__ counts) {
  const long long idx = (long long)blockIdx.x * blockDim.x + threadIdx.x;
  if (idx >= (long long)NNODES * CH) return;
  out_sum[idx] /= fmaxf(counts[idx >> 6], 1.0f);
}

// ---------------------------------------------------------------------------
extern "C" void kernel_launch(void* const* d_in, const int* in_sizes, int n_in,
                              void* d_out, int out_size, void* d_ws, size_t ws_size,
                              hipStream_t stream) {
  const float*     x     = (const float*)d_in[0];
  const long long* ei    = (const long long*)d_in[1];
  const float*     W0    = (const float*)d_in[2];
  const float*     b0    = (const float*)d_in[3];
  const float*     g0    = (const float*)d_in[4];
  const float*     beta0 = (const float*)d_in[5];
  const float*     W1    = (const float*)d_in[6];
  const float*     b1    = (const float*)d_in[7];
  const float*     g1    = (const float*)d_in[8];
  const float*     beta1 = (const float*)d_in[9];
  const float*     W2    = (const float*)d_in[10];
  const float*     b2    = (const float*)d_in[11];
  const float*     g2    = (const float*)d_in[12];
  const float*     beta2 = (const float*)d_in[13];

  float* out_sum  = (float*)d_out;                       // [NNODES*CH]
  float* edge_act = out_sum + (long long)NNODES * CH;    // [NE*CH]
  unsigned char* ws = (unsigned char*)d_ws;
  float* stats  = (float*)(ws + WS_STATS);
  float* counts = (float*)(ws + WS_CNT);
  _Float16* h0 = (_Float16*)(ws + WS_H0);
  _Float16* h1 = (_Float16*)(ws + WS_H1);
  _Float16* h2 = h0;                                     // h0 dead after layer 1
  const _Float16* w1t = (const _Float16*)(ws + WS_W1T);
  const _Float16* w2t = (const _Float16*)(ws + WS_W2T);

  k_init<<<2048, 256, 0, stream>>>(W0, W1, W2, out_sum, ws);

  k_edge_gemm0<<<NE / 64, 128, 0, stream>>>(x, ei, b0, ws);
  k_bn_finalize<<<1, 64, 0, stream>>>(stats + ST_SUM0, stats + ST_SQ0, g0, beta0,
                                      stats + ST_SC0, stats + ST_SH0);

  k_edge_gemm_mid<<<NE / 64, 128, 0, stream>>>(h0, h1, w1t, b1,
                                               stats + ST_SC0, stats + ST_SH0,
                                               stats + ST_SUM1, stats + ST_SQ1);
  k_bn_finalize<<<1, 64, 0, stream>>>(stats + ST_SUM1, stats + ST_SQ1, g1, beta1,
                                      stats + ST_SC1, stats + ST_SH1);

  k_edge_gemm_mid<<<NE / 64, 128, 0, stream>>>(h1, h2, w2t, b2,
                                               stats + ST_SC1, stats + ST_SH1,
                                               stats + ST_SUM2, stats + ST_SQ2);
  k_bn_finalize<<<1, 64, 0, stream>>>(stats + ST_SUM2, stats + ST_SQ2, g2, beta2,
                                      stats + ST_SC2, stats + ST_SH2);

  const long long tot = (long long)NE * CH;
  k_edge_final<<<(int)((tot + 255) / 256), 256, 0, stream>>>(
      h2, ei, stats + ST_SC2, stats + ST_SH2, out_sum, edge_act, counts);

  k_node_div<<<(NNODES * CH + 255) / 256, 256, 0, stream>>>(out_sum, counts);
}
